// MotifPredictor_22247930593973
// MI455X (gfx1250) — matrix operand order repaired and checked by reading, not compiled, not run on hardware
//
#include <hip/hip_runtime.h>

// ---------------------------------------------------------------------------
// MotifPredictor for MI455X (gfx1250, wave32, WMMA + TDM).
//
// Roofline: union_features (256MB f32) is the mandatory HBM traffic (~12us at
// 23.3 TB/s) -> streamed with non-temporal loads. Dominant GEMM is
// 16384x1024x4096 = 137 GFLOP -> bf16 WMMA (v_wmma_f32_16x16x32_bf16, f32
// accumulate). The 16384x4096 intermediate stays in registers/LDS, fused with
// the x-union epilogue and the final 4096->51 GEMM (padded to 64 cols),
// accumulated via f32 atomics across 8 p-slices. The stage-1 B tile is DMA'd
// global->LDS by the Tensor Data Mover (tensor_load_to_lds, double-buffered,
// TENSORcnt-waited) with TDM pad_enable producing a 272B row stride so the
// WMMA B-fragment ds_load_b128 reads are bank-conflict free.
// ---------------------------------------------------------------------------

typedef __bf16 bf16;
typedef __attribute__((ext_vector_type(16))) bf16  bf16x16;
typedef __attribute__((ext_vector_type(8)))  bf16  bf16x8;
typedef __attribute__((ext_vector_type(8)))  float floatx8;
typedef unsigned int u32;
typedef __attribute__((ext_vector_type(4))) u32 u32x4;
typedef __attribute__((ext_vector_type(8))) u32 u32x8;

#define HID         512
#define POOL        4096
#define N_OBJ       1280
#define N_REL       16384
#define NUM_OBJ_CLS 151
#define NUM_REL_CLS 51
#define RELPAD      64      // pad 51 -> 64 (4 WMMA n-subtiles)

// Build a 16-element bf16 WMMA fragment from two contiguous 8-element chunks.
__device__ __forceinline__ bf16x16 make_frag(const bf16* p0, const bf16* p1) {
    bf16x8 lo = *(const bf16x8*)p0;
    bf16x8 hi = *(const bf16x8*)p1;
    return __builtin_shufflevector(lo, hi, 0,1,2,3,4,5,6,7,8,9,10,11,12,13,14,15);
}

// ---------------------------------------------------------------------------
// TDM: DMA a 128(n) x 128(k) bf16 tile of WtCat[n][1024] into LDS at ldsOff.
// D# per CDNA5 ISA (group0 128b, group1 256b):
//   group0: count=1 | lds_addr | global_addr(57b) | type=2
//   group1: data_size=2B, pad_enable, pad_interval=64 DWORDs (code 5),
//           pad_amount=4 DWORDs (code 3)  -> LDS row stride 256+16=272B,
//           tensor_dim0=1024, tensor_dim1=4096, tile_dim0=128, tile_dim1=128,
//           tensor_dim0_stride=1024, workgroup_mask=0 (not in cluster).
// Issued by one wave only (TDM is wave-scalar, EXEC-ignored); completion via
// s_wait_tensorcnt then workgroup barrier.
// ---------------------------------------------------------------------------
__device__ __forceinline__ void tdm_load_Btile(const bf16* gsrc, unsigned ldsOff) {
    unsigned long long ga = (unsigned long long)(size_t)gsrc;
    u32x4 g0;
    g0.x = 1u;                                   // count=1 (valid descriptor)
    g0.y = ldsOff;                               // lds_addr
    g0.z = (u32)ga;                              // global_addr[31:0]
    g0.w = ((u32)(ga >> 32) & 0x01FFFFFFu) | (2u << 30);   // addr[56:32] | type=2
    u32x8 g1 = { (1u << 16) | (1u << 20) | (5u << 22) | (3u << 25),
                 1024u << 16,      // tensor_dim0 = 1024 (bits 79:48)
                 4096u << 16,      // tensor_dim1 = 4096 (bits 111:80)
                 128u  << 16,      // tile_dim0 = 128 (bits 127:112)
                 128u,             // tile_dim1 = 128 (bits 143:128)
                 1024u,            // tensor_dim0_stride = 1024 (bits 207:160)
                 0u, 0u };
    asm volatile("tensor_load_to_lds %0, %1, null, null" :: "s"(g0), "s"(g1) : "memory");
}

// --------------------------- f32 -> bf16 convert ---------------------------
__global__ void k_conv_bf16(const float* __restrict__ src, bf16* __restrict__ dst, int n) {
    int i = blockIdx.x * 256 + threadIdx.x;
    if (i < n) dst[i] = (bf16)src[i];
}

// ---------------- tiled transpose + convert: src[K][N]f32 -> dst[Npad][K]bf16
__global__ void k_transpose_bf16(const float* __restrict__ src, bf16* __restrict__ dst,
                                 int K, int N, int Npad) {
    __shared__ float t[32][33];
    const int kb = blockIdx.y * 32, nb = blockIdx.x * 32;
    const int tx = threadIdx.x, ty = threadIdx.y;          // 32 x 8
    for (int j = 0; j < 32; j += 8) {
        int k = kb + ty + j, n = nb + tx;
        t[ty + j][tx] = (k < K && n < N) ? src[(long)k * N + n] : 0.0f;
    }
    __syncthreads();
    for (int j = 0; j < 32; j += 8) {
        int n = nb + ty + j, k = kb + tx;
        if (n < Npad && k < K) dst[(long)n * K + k] = (bf16)t[tx][ty + j];
    }
}

// ------------- edge_rep = edge_ctx @ W_post_emb + b  (bf16 out) -------------
// A: ectx_bf [N_OBJ][HID] bf16 ; Bt: WtEmb [1024][HID] bf16 (n-major)
__global__ void __launch_bounds__(256)
k_edge_gemm(const bf16* __restrict__ A, const bf16* __restrict__ Bt,
            const float* __restrict__ bias, bf16* __restrict__ out) {
    const int wave = threadIdx.x >> 5, lane = threadIdx.x & 31;
    const int half = lane >> 4, mr = lane & 15;
    const int rowBase = blockIdx.x * 128 + wave * 16;      // 10 * 128 = 1280
    const int nBase   = blockIdx.y * 128;                  // 8 * 128 = 1024

    floatx8 acc[8] = {};
    const bf16* arow = A + (long)(rowBase + mr) * HID;
    for (int kb = 0; kb < HID; kb += 32) {
        bf16x16 a = make_frag(arow + kb + half * 16, arow + kb + half * 16 + 8);
        for (int s = 0; s < 8; ++s) {
            const bf16* bp = Bt + (long)(nBase + s * 16 + mr) * HID + kb + half * 16;
            bf16x16 b = make_frag(bp, bp + 8);
            acc[s] = __builtin_amdgcn_wmma_f32_16x16x32_bf16(
                false, a, false, b, (short)0, acc[s], false, false);
        }
    }
    for (int s = 0; s < 8; ++s) {
        const int c = nBase + s * 16 + mr;
        const float bc = bias[c];
        for (int i = 0; i < 8; ++i) {
            const int r = rowBase + i + half * 8;          // C/D layout (ISA 7.12.2)
            out[(long)r * 1024 + c] = (bf16)(acc[s][i] + bc);
        }
    }
}

// ---------------- out[r][c] = b_rel[c] + freq_table[bias_idx[r]][c] ---------
__global__ void k_init_out(const float* __restrict__ b_rel, const float* __restrict__ freq,
                           const int* __restrict__ pair, const int* __restrict__ obj,
                           float* __restrict__ out) {
    const int r = blockIdx.x, c = threadIdx.x;
    if (c >= NUM_REL_CLS) return;
    const long bi = (long)obj[pair[r * 2]] * NUM_OBJ_CLS + obj[pair[r * 2 + 1]];
    out[(long)r * NUM_REL_CLS + c] = b_rel[c] + freq[bi * NUM_REL_CLS + c];
}

// --------------------------- fused main kernel -----------------------------
// grid (128 Mtiles, 8 p-slices of 512). Block = 256 = 8 waves.
// stage1: gathered 128x128 tile of (prod_rep_in @ W_post_cat), K=1024, WMMA;
//         B tiles DMA'd by TDM (double-buffered, 128-wide K chunks).
// epilogue: (+b_cat) * union (NT loads) -> bf16 LDS tile Vs.
// stage2: Vs(128 x 128p) @ WtRel -> 128x64 partial, accumulated over 4 nt,
//         then global_atomic_add_f32 into out (c < 51).
__global__ void __launch_bounds__(256)
k_fused(const bf16* __restrict__ edge_rep,   // [N_OBJ][1024] bf16
        const bf16* __restrict__ WtCat,      // [POOL][1024]  bf16 (n-major)
        const bf16* __restrict__ WtRel,      // [RELPAD][POOL] bf16 (n-major, zero pad)
        const float* __restrict__ b_cat,     // [POOL]
        const float* __restrict__ uni,       // [N_REL][POOL] f32
        const int*   __restrict__ pair,      // [N_REL][2]
        float*       __restrict__ out)       // [N_REL][51] accumulate
{
    __shared__ bf16 Bs[2][128][136];   // TDM-padded B tiles [n][k], row = 272B
    __shared__ bf16 Vs[128][136];      // fused intermediate [r][p], row = 272B

    const int wave = threadIdx.x >> 5, lane = threadIdx.x & 31;
    const int half = lane >> 4, mr = lane & 15;
    const int mBase = blockIdx.x * 128;
    const int sliceBase = blockIdx.y * 512;

    // flat-address low 32 bits == LDS byte offset (ISA 10.2 aperture rules)
    const unsigned ldsB0 = (unsigned)(size_t)(void*)&Bs[0][0][0];
    const unsigned ldsB1 = (unsigned)(size_t)(void*)&Bs[1][0][0];

    // Gathered A rows: prod_rep_in[r][k] = k<512 ? edge_rep[p0][k] : edge_rep[p1][k]
    const int rA = mBase + wave * 16 + mr;
    const int p0 = pair[rA * 2 + 0];
    const int p1 = pair[rA * 2 + 1];
    const bf16* rowH = edge_rep + (long)p0 * 1024;
    const bf16* rowT = edge_rep + (long)p1 * 1024;

    floatx8 acc2[4] = {};

    for (int nt = 0; nt < 4; ++nt) {
        const int pBase = sliceBase + nt * 128;
        __builtin_prefetch(&uni[(long)rA * POOL + pBase], 0, 0);  // global_prefetch_b8

        floatx8 acc1[8] = {};

        if (wave == 0)                       // prime the TDM pipeline
            tdm_load_Btile(WtCat + (long)pBase * 1024, ldsB0);

        for (int kc = 0; kc < 8; ++kc) {     // 128-wide K chunks
            const int buf = kc & 1;
            if (wave == 0) {
                if (kc + 1 < 8) {
                    tdm_load_Btile(WtCat + (long)pBase * 1024 + (kc + 1) * 128,
                                   ((kc + 1) & 1) ? ldsB1 : ldsB0);
                    __builtin_amdgcn_s_wait_tensorcnt(1);   // chunk kc landed
                } else {
                    __builtin_amdgcn_s_wait_tensorcnt(0);
                }
            }
            __syncthreads();                 // Bs[buf] visible to all waves

            for (int kbl = 0; kbl < 128; kbl += 32) {
                const int kg = kc * 128 + kbl;
                const bf16* arow = (kg < 512) ? rowH : rowT;
                bf16x16 a = make_frag(arow + kg + half * 16, arow + kg + half * 16 + 8);
                for (int s = 0; s < 8; ++s) {
                    const bf16* bp = &Bs[buf][s * 16 + mr][kbl + half * 16];
                    bf16x16 b = make_frag(bp, bp + 8);
                    acc1[s] = __builtin_amdgcn_wmma_f32_16x16x32_bf16(
                        false, a, false, b, (short)0, acc1[s], false, false);
                }
            }
            __syncthreads();                 // done reading Bs[buf] before reuse
        }

        // epilogue: v = (acc1 + b_cat[p]) * union[r][p]  -> Vs (bf16)
        for (int s = 0; s < 8; ++s) {
            const int p = pBase + s * 16 + mr;
            const float bc = b_cat[p];
            for (int i = 0; i < 8; ++i) {
                const int m = i + half * 8;
                const int r = mBase + wave * 16 + m;
                const float u = __builtin_nontemporal_load(&uni[(long)r * POOL + p]);
                Vs[wave * 16 + m][s * 16 + mr] = (bf16)((acc1[s][i] + bc) * u);
            }
        }
        __syncthreads();

        // stage-2: acc2 += Vs(wave's 16 rows x 128 p) @ WtRel(p x 64)
        const bf16* vrow = &Vs[wave * 16 + mr][0];
        for (int kb = 0; kb < 128; kb += 32) {
            bf16x16 a = make_frag(vrow + kb + half * 16, vrow + kb + half * 16 + 8);
            for (int s = 0; s < 4; ++s) {
                const bf16* bp = WtRel + (long)(s * 16 + mr) * POOL + pBase + kb + half * 16;
                bf16x16 b = make_frag(bp, bp + 8);
                acc2[s] = __builtin_amdgcn_wmma_f32_16x16x32_bf16(
                    false, a, false, b, (short)0, acc2[s], false, false);
            }
        }
        __syncthreads();                     // Vs reads done before next nt writes
    }

    // accumulate this p-slice's partial rel_dists
    for (int s = 0; s < 4; ++s) {
        const int c = s * 16 + mr;
        if (c < NUM_REL_CLS) {
            for (int i = 0; i < 8; ++i) {
                const int r = mBase + wave * 16 + i + half * 8;
                atomicAdd(&out[(long)r * NUM_REL_CLS + c], acc2[s][i]);
            }
        }
    }
}

// ---------------------------------------------------------------------------
extern "C" void kernel_launch(void* const* d_in, const int* in_sizes, int n_in,
                              void* d_out, int out_size, void* d_ws, size_t ws_size,
                              hipStream_t stream) {
    const float* edge_ctx       = (const float*)d_in[0];
    const float* union_features = (const float*)d_in[1];
    const float* W_post_emb     = (const float*)d_in[2];
    const float* b_post_emb     = (const float*)d_in[3];
    const float* W_post_cat     = (const float*)d_in[4];
    const float* b_post_cat     = (const float*)d_in[5];
    const float* W_rel          = (const float*)d_in[6];
    const float* b_rel          = (const float*)d_in[7];
    const float* freq_table     = (const float*)d_in[8];
    const int*   pair_idx       = (const int*)d_in[9];
    const int*   obj_preds      = (const int*)d_in[10];
    float* out = (float*)d_out;

    // workspace layout (~13.9 MB total)
    char* ws = (char*)d_ws;
    size_t off = 0;
    bf16* ectx_bf = (bf16*)(ws + off); off += (size_t)N_OBJ * HID * 2;
    bf16* WtEmb   = (bf16*)(ws + off); off += (size_t)1024 * HID * 2;
    bf16* WtCat   = (bf16*)(ws + off); off += (size_t)POOL * 1024 * 2;
    bf16* WtRel   = (bf16*)(ws + off); off += (size_t)RELPAD * POOL * 2;
    bf16* erep    = (bf16*)(ws + off); off += (size_t)N_OBJ * 1024 * 2;
    (void)ws_size; (void)in_sizes; (void)n_in; (void)out_size;

    dim3 tb(32, 8);
    k_conv_bf16<<<(N_OBJ * HID + 255) / 256, 256, 0, stream>>>(edge_ctx, ectx_bf, N_OBJ * HID);
    k_transpose_bf16<<<dim3(1024 / 32, HID / 32), tb, 0, stream>>>(W_post_emb, WtEmb, HID, 1024, 1024);
    k_transpose_bf16<<<dim3(POOL / 32, 1024 / 32), tb, 0, stream>>>(W_post_cat, WtCat, 1024, POOL, POOL);
    k_transpose_bf16<<<dim3(RELPAD / 32, POOL / 32), tb, 0, stream>>>(W_rel, WtRel, POOL, NUM_REL_CLS, RELPAD);

    k_edge_gemm<<<dim3(N_OBJ / 128, 1024 / 128), 256, 0, stream>>>(ectx_bf, WtEmb, b_post_emb, erep);
    k_init_out<<<N_REL, 64, 0, stream>>>(b_rel, freq_table, pair_idx, obj_preds, out);
    k_fused<<<dim3(N_REL / 128, 8), 256, 0, stream>>>(erep, WtCat, WtRel, b_post_cat,
                                                      union_features, pair_idx, out);
}
